// MLP_two_26757646254173
// MI455X (gfx1250) — matrix operand order repaired and checked by reading, not compiled, not run on hardware
//
#include <hip/hip_runtime.h>
#include <hip/hip_bf16.h>
#include <math.h>

typedef __attribute__((ext_vector_type(16))) _Float16 v16h;
typedef __attribute__((ext_vector_type(8)))  float    v8f;

#define B_      4096
#define HN_     16
#define N1_     192
#define DIM_    256
#define TWODIM  512
#define EPS_    1e-5f

#define NB      4              // batches per workgroup (4 M-tiles of 16 rows)
#define ROWS    (NB * HN_)     // 64 rows per workgroup
#define THREADS 256            // 8 wave32

// ---------------------------------------------------------------------------
// Pack W1 (256x512 f32) and W2 (512x256 f32) into f16 WMMA B-operand layout.
// B fragment as packed here for v_wmma_f32_16x16x32_f16 (wave32, contiguous K, same order as A):
//   lane = (K-half<<4) | N16 ;  lanes 0-15 hold K=0..15, lanes 16-31 K=16..31
//   the 16 halves of the lane's v16h are K ascending within that half-block.
// Linear layout: W1p[(((nt*KI)+ki)*32 + lane)*16 + j]  -> one v16h = 32B/lane.
// ---------------------------------------------------------------------------
__global__ __launch_bounds__(256)
void pack_weights(const float* __restrict__ W1, const float* __restrict__ W2,
                  _Float16* __restrict__ W1p, _Float16* __restrict__ W2p) {
  int tid = blockIdx.x * blockDim.x + threadIdx.x;      // 0 .. 262143
  int j    = tid & 15;
  int lane = (tid >> 4) & 31;
  int n16  = lane & 15;
  int half = lane >> 4;
  if (tid < DIM_ * TWODIM) {                            // W1: N=256, K=512, KI=16
    int rest = tid >> 9;                                // nt*16 + ki
    int ki = rest & 15, nt = rest >> 4;
    int n = nt * 16 + n16;                              // 0..255 (d)
    int k = ki * 32 + 16 * half + j;                    // 0..511 (e)
    W1p[tid] = (_Float16)W1[n * TWODIM + k];
  } else {                                              // W2: N=512, K=256, KI=8
    int t = tid - DIM_ * TWODIM;
    int rest = t >> 9;                                  // nt*8 + ki
    int ki = rest & 7, nt = rest >> 3;
    int n = nt * 16 + n16;                              // 0..511 (e)
    int k = ki * 32 + 16 * half + j;                    // 0..255 (d)
    W2p[t] = (_Float16)W2[n * DIM_ + k];
  }
}

// ---------------------------------------------------------------------------
// Fused scatter + LayerNorm + MLP(512->256 relu, 256->512 sigmoid) + gather.
// One workgroup = 4 consecutive batches = 64 rows (4 WMMA M-tiles).
// ---------------------------------------------------------------------------
__global__ __launch_bounds__(THREADS)
void mlp_fused(const float* __restrict__ rgb, const float* __restrict__ tir,
               const int* __restrict__ gidx,
               const float* __restrict__ ln_g, const float* __restrict__ ln_b,
               const float* __restrict__ b1,   const float* __restrict__ b2,
               const _Float16* __restrict__ W1p, const _Float16* __restrict__ W2p,
               float* __restrict__ out_rgb, float* __restrict__ out_tir) {
  extern __shared__ char smem[];
  // Region 0 (128 KB): X (f16 [64][512]) in phases A/B; O (f32 [64][512]) in C/D.
  _Float16* Xs   = (_Float16*)smem;
  float*    Os   = (float*)smem;
  _Float16* Hs   = (_Float16*)(smem + 131072);          // f16 [64][256] = 32 KB
  int*      idxs = (int*)(smem + 131072 + 32768);       // [4][192]

  const int tid  = threadIdx.x;
  const int b0   = blockIdx.x * NB;                     // first batch of this WG
  const int row  = tid >> 2;                            // 0..63  (= bb*16 + h)
  const int sub  = tid & 3;                             // 4 threads per row
  const int grow = b0 * HN_ + row;                      // global (b,h) row

  // idx -> LDS (4 batches contiguous in gidx)
  for (int k = tid; k < NB * N1_; k += THREADS)
    idxs[k] = gidx[(size_t)b0 * N1_ + k];

  // ---- Phase A1: LN stats directly from the 384 nonzeros (rest are zero) ----
  const float* rrow = rgb + (size_t)grow * N1_;
  const float* trow = tir + (size_t)grow * N1_;
  float s = 0.f, ss = 0.f;
  for (int i = sub; i < N1_; i += 4) {
    float a = rrow[i], c = trow[i];
    s += a + c;  ss += a * a + c * c;
  }
  s  += __shfl_xor(s, 1);  s  += __shfl_xor(s, 2);      // reduce the 4 row-lanes
  ss += __shfl_xor(ss, 1); ss += __shfl_xor(ss, 2);
  const float mu = s * (1.0f / 512.0f);
  const float rs = rsqrtf(ss * (1.0f / 512.0f) - mu * mu + EPS_);

  // ---- Phase A2: default fill X = (0-mu)*rs*g + b, 8 halves per b128 store --
  {
    const float nm = -mu * rs;
    for (int c = sub * 128; c < sub * 128 + 128; c += 8) {
      union { _Float16 h[8]; uint4 u; } pk;
#pragma unroll
      for (int q = 0; q < 8; ++q)
        pk.h[q] = (_Float16)(nm * ln_g[c + q] + ln_b[c + q]);
      *(uint4*)(Xs + row * TWODIM + c) = pk.u;
    }
  }
  __syncthreads();

  // ---- Phase A3: scatter-overwrite the indexed slots --------------------
  {
    const int* ir = idxs + (row >> 4) * N1_;
    for (int i = sub; i < N1_; i += 4) {
      int c = ir[i];
      float xr = (rrow[i] - mu) * rs;
      float xt = (trow[i] - mu) * rs;
      Xs[row * TWODIM + c]        = (_Float16)(xr * ln_g[c]        + ln_b[c]);
      Xs[row * TWODIM + 256 + c]  = (_Float16)(xt * ln_g[256 + c]  + ln_b[256 + c]);
    }
  }
  __syncthreads();

  // ---- Phase B: GEMM1  H[64x256] = relu(X[64x512] * W1^T + b1) ----------
  const int lane = tid & 31;
  const int wave = tid >> 5;          // 0..7
  const int nlo  = lane & 15;
  const int half = lane >> 4;

  for (int tl = 0; tl < 2; ++tl) {
    const int nt = wave + tl * 8;     // 16 N-tiles over 8 waves
    v8f acc[4] = {};
    const v16h* bp = (const v16h*)W1p + (size_t)(nt * 16) * 32 + lane;
    for (int ki = 0; ki < 16; ++ki) {
      v16h bm = bp[ki * 32];          // 32B/lane, L2-resident
#pragma unroll
      for (int mt = 0; mt < 4; ++mt) {
        // A layout: lane row = nlo; K = ki*32 + 16*half + {0..15}
        const uint4* xp = (const uint4*)(Xs + (mt * 16 + nlo) * TWODIM + ki * 32 + 16 * half);
        union { uint4 u[2]; v16h h; } am;
        am.u[0] = xp[0];
        am.u[1] = xp[1];              // +8 halves
        acc[mt] = __builtin_amdgcn_wmma_f32_16x16x32_f16(
            false, am.h, false, bm, (short)0, acc[mt], false, false);
      }
    }
    const float bias = b1[nt * 16 + nlo];
#pragma unroll
    for (int mt = 0; mt < 4; ++mt)
#pragma unroll
      for (int r = 0; r < 8; ++r) {
        float v = acc[mt][r] + bias;
        v = v > 0.f ? v : 0.f;
        Hs[(mt * 16 + r + 8 * half) * DIM_ + nt * 16 + nlo] = (_Float16)v;
      }
  }
  __syncthreads();

  // ---- Phase C: GEMM2  O[64x512] = sigmoid(H[64x256] * W2^T + b2) -------
  for (int tl = 0; tl < 4; ++tl) {
    const int nt = wave + tl * 8;     // 32 N-tiles over 8 waves
    v8f acc[4] = {};
    const v16h* bp = (const v16h*)W2p + (size_t)(nt * 8) * 32 + lane;
    for (int ki = 0; ki < 8; ++ki) {
      v16h bm = bp[ki * 32];
#pragma unroll
      for (int mt = 0; mt < 4; ++mt) {
        const uint4* hp = (const uint4*)(Hs + (mt * 16 + nlo) * DIM_ + ki * 32 + 16 * half);
        union { uint4 u[2]; v16h h; } am;
        am.u[0] = hp[0];
        am.u[1] = hp[1];
        acc[mt] = __builtin_amdgcn_wmma_f32_16x16x32_f16(
            false, am.h, false, bm, (short)0, acc[mt], false, false);
      }
    }
    const float bias = b2[nt * 16 + nlo];
#pragma unroll
    for (int mt = 0; mt < 4; ++mt)
#pragma unroll
      for (int r = 0; r < 8; ++r) {
        float v = acc[mt][r] + bias;
        v = 1.0f / (1.0f + __expf(-v));
        Os[(mt * 16 + r + 8 * half) * TWODIM + nt * 16 + nlo] = v;
      }
  }
  __syncthreads();

  // ---- Phase D: gather through idx, coalesced stores --------------------
  {
    const int* ir = idxs + (row >> 4) * N1_;
    const float* orow = Os + row * TWODIM;
    float* og = out_rgb + (size_t)grow * N1_;
    float* ot = out_tir + (size_t)grow * N1_;
    for (int i = sub; i < N1_; i += 4) {
      int c = ir[i];
      og[i] = orow[c];
      ot[i] = orow[256 + c];
    }
  }
}

// ---------------------------------------------------------------------------
extern "C" void kernel_launch(void* const* d_in, const int* in_sizes, int n_in,
                              void* d_out, int out_size, void* d_ws, size_t ws_size,
                              hipStream_t stream) {
  const float* rgb  = (const float*)d_in[0];
  const float* tir  = (const float*)d_in[1];
  const int*   gidx = (const int*)d_in[2];
  const float* ln_g = (const float*)d_in[3];
  const float* ln_b = (const float*)d_in[4];
  const float* W1   = (const float*)d_in[5];
  const float* b1   = (const float*)d_in[6];
  const float* W2   = (const float*)d_in[7];
  const float* b2   = (const float*)d_in[8];

  _Float16* W1p = (_Float16*)d_ws;                 // 256 KB
  _Float16* W2p = W1p + DIM_ * TWODIM;             // 256 KB

  pack_weights<<<(2 * DIM_ * TWODIM) / 256, 256, 0, stream>>>(W1, W2, W1p, W2p);

  float* out_rgb = (float*)d_out;
  float* out_tir = out_rgb + (size_t)B_ * HN_ * N1_;

  size_t shmem = 131072 + 32768 + NB * N1_ * sizeof(int);   // ~164 KB of 320 KB
  mlp_fused<<<B_ / NB, THREADS, shmem, stream>>>(
      rgb, tir, gidx, ln_g, ln_b, b1, b2, W1p, W2p, out_rgb, out_tir);
}
